// _HGN_31155692765823
// MI455X (gfx1250) — matrix operand order repaired and checked by reading, not compiled, not run on hardware
//
#include <hip/hip_runtime.h>

// ---------------------------------------------------------------------------
// HGN-style user-representation + scoring, CDNA5 (gfx1250, wave32, WMMA+TDM).
//
// res[b,t] = b2[id] + dot(w2[b,t], v[b])
//   v[b]   = user_emb[b] + union[b] + sum_l item_embs[b,l]
// ---------------------------------------------------------------------------

#if defined(__HIP_DEVICE_COMPILE__)
#if __has_include(<hip/amd_detail/amd_gfx1250_TDM.h>)
#include <hip/amd_detail/amd_gfx1250_TDM.h>   // therock-10.0 toolchain marker
#endif
#endif

typedef __attribute__((ext_vector_type(16))) _Float16 v16h;
typedef __attribute__((ext_vector_type(8)))  float    v8f;
typedef __attribute__((ext_vector_type(4)))  unsigned int u32x4;
typedef __attribute__((ext_vector_type(8)))  int      i32x8;
typedef __attribute__((ext_vector_type(4)))  int      i32x4;

#define DD 64
#define LP 68   // padded fp32 row stride in LDS (floats); 68*4B rows
#define WP 66   // padded f16 weight row stride (halves): 66*2B = 33 banks, gcd(33,64)=1

#if defined(__HIP_DEVICE_COMPILE__)
#if __has_builtin(__builtin_amdgcn_tensor_load_to_lds) && __has_builtin(__builtin_amdgcn_s_wait_tensorcnt)
#define USE_TDM_GATHER 1
#endif
#endif

#ifdef USE_TDM_GATHER
__device__ __forceinline__ void tdm_gather_load(u32x4 g0, i32x8 g1, i32x4 g2, i32x4 g3) {
#if __clang_major__ >= 23
    // amdgpu-toolchain (clang-23 / therock-10.0): 6-arg form, extra zero-filled i32x8.
    i32x8 z8 = {};
    __builtin_amdgcn_tensor_load_to_lds(g0, g1, g2, g3, z8, 0);
#else
    // ROCm 7.2 (clang-22): 5-arg form.
    __builtin_amdgcn_tensor_load_to_lds(g0, g1, g2, g3, 0);
#endif
}
#endif

__device__ __forceinline__ float sigmoidf(float x) {
    return 1.0f / (1.0f + __expf(-x));
}

// One block per batch row b. 128 threads = 4 wave32.
__global__ __launch_bounds__(128) void hgn_user_kernel(
    const int*   __restrict__ user_ids,      // [B]
    const int*   __restrict__ item_seq_ids,  // [B,L]
    const float* __restrict__ user_table,    // [U,64]
    const float* __restrict__ item_table,    // [I,64]
    const float* __restrict__ fg_item_W,     // [64,64]
    const float* __restrict__ fg_item_b,     // [64]
    const float* __restrict__ fg_user_W,     // [64,64]
    const float* __restrict__ fg_user_b,     // [64]
    const float* __restrict__ inst_gate_item,// [64,1]
    const float* __restrict__ inst_gate_user,// [64,L]
    float*       __restrict__ v_out,         // [B,64]
    int L, int num_items)
{
    __shared__ __align__(16) float    items_s[64 * LP];   // gathered item embs (rows >= L zero)
    __shared__ __align__(16) float    gated_s[64 * LP];   // item * sigmoid(gate)
    __shared__ __align__(16) _Float16 wf16_s[64 * WP];    // fg_item_W in f16
    __shared__ float user_s[64];
    __shared__ float gadd_s[64];                          // per-column additive bias
    __shared__ float inst_s[64];

    const int b    = blockIdx.x;
    const int tid  = threadIdx.x;
    const int lane = tid & 31;
    const int wave = tid >> 5;     // 0..3 -> M tile

    // -------- stage user embedding --------
    if (tid < 64) {
        user_s[tid] = user_table[(size_t)user_ids[b] * DD + tid];
    }

    // -------- gather item embeddings into LDS (padded 64 rows) --------
#ifdef USE_TDM_GATHER
    // Zero the pad rows L..63 with regular threads while wave 0 drives the TDM.
    for (int idx = tid; idx < (64 - L) * 16; idx += 128) {
        int row = L + (idx >> 4), q = idx & 15;
        *(float4*)(&items_s[row * LP + q * 4]) = make_float4(0.f, 0.f, 0.f, 0.f);
    }
    if (wave == 0) {
        // Tensor Data Mover, gather mode: 8 x 32-bit row indices per descriptor,
        // each row = 64 fp32; LDS padding 4 DWORDs after every 64 DWORDs keeps
        // the stride-68 bank-conflict-free layout.
        const unsigned long long ga = (unsigned long long)item_table;
        const unsigned ldsbase = (unsigned)(unsigned long long)(&items_s[0]);
        const int ndesc = (L + 7) >> 3;
        for (int dsc = 0; dsc < ndesc; ++dsc) {
            int nrows = L - dsc * 8; if (nrows > 8) nrows = 8;
            int idxv[8];
            #pragma unroll
            for (int j = 0; j < 8; ++j) {
                int r = dsc * 8 + j;
                idxv[j] = (j < nrows)
                    ? __builtin_amdgcn_readfirstlane(item_seq_ids[(size_t)b * L + r])
                    : 0;
            }
            u32x4 g0; i32x8 g1; i32x4 g2, g3;
            // Group 0: count=1, gather_mode=1, 32-bit indices, lds_addr, global_addr, type=2
            g0[0] = 0xC0000001u;
            g0[1] = ldsbase + (unsigned)(dsc * 8 * LP * 4);
            g0[2] = (unsigned)(ga & 0xFFFFFFFFull);
            g0[3] = (unsigned)((ga >> 32) & 0x1FFFFFFull) | (2u << 30);
            // Group 1: data_size=4B, pad_enable, pad_interval=64dw(code5), pad_amount=4dw(code3)
            g1[0] = (2 << 16) | (1 << 20) | (5 << 22) | (3 << 25);
            g1[1] = (64 << 16);                                     // tensor_dim0 = 64
            g1[2] = (int)(((unsigned)num_items & 0xFFFFu) << 16);   // tensor_dim1 lo
            g1[3] = (int)((((unsigned)num_items >> 16) & 0xFFFFu) | (64u << 16)); // dim1 hi | tile_dim0=64
            g1[4] = nrows;                                          // tile_dim1 = #valid indices
            g1[5] = 64;                                             // tensor_dim0_stride = 64
            g1[6] = 0;
            g1[7] = 0;
            // Groups 2/3: 32-bit row indices
            g2[0] = idxv[0]; g2[1] = idxv[1]; g2[2] = idxv[2]; g2[3] = idxv[3];
            g3[0] = idxv[4]; g3[1] = idxv[5]; g3[2] = idxv[6]; g3[3] = idxv[7];
            tdm_gather_load(g0, g1, g2, g3);
        }
        __builtin_amdgcn_s_wait_tensorcnt(0);
    }
#else
    for (int idx = tid; idx < 64 * 16; idx += 128) {
        int row = idx >> 4, q = idx & 15;
        float4 val;
        if (row < L) {
            size_t src = (size_t)item_seq_ids[(size_t)b * L + row] * DD;
            val = *(const float4*)(item_table + src + q * 4);
        } else {
            val = make_float4(0.f, 0.f, 0.f, 0.f);
        }
        *(float4*)(&items_s[row * LP + q * 4]) = val;
    }
#endif

    // -------- stage fg_item_W as f16 --------
    for (int idx = tid; idx < 64 * 64; idx += 128) {
        int row = idx >> 6, col = idx & 63;
        wf16_s[row * WP + col] = (_Float16)fg_item_W[idx];
    }
    __syncthreads();

    // -------- per-column bias: fg_item_b + fg_user_b + user @ fg_user_W^T --------
    if (tid < 64) {
        float acc = fg_item_b[tid] + fg_user_b[tid];
        #pragma unroll 8
        for (int d = 0; d < 64; ++d) acc += user_s[d] * fg_user_W[tid * 64 + d];
        gadd_s[tid] = acc;
    }
    __syncthreads();

    // -------- WMMA: S = items(64x64) @ fg_item_W^T  (f16 in, f32 acc) --------
    v8f acc[4] = {};
    const int mrow  = wave * 16 + (lane & 15);
    const int khalf = (lane >> 4) << 3;     // A-frag K interleave (0 or 8)
    const int koffB = (lane >> 4) << 3;     // B-frag K interleave (0 or 8)

    #pragma unroll
    for (int ks = 0; ks < 2; ++ks) {
        v16h afrag;
        #pragma unroll
        for (int v = 0; v < 8; ++v) {
            int kb = (v < 4) ? (v * 2) : (16 + (v - 4) * 2);
            int k  = ks * 32 + khalf + kb;
            afrag[2 * v]     = (_Float16)items_s[mrow * LP + k];
            afrag[2 * v + 1] = (_Float16)items_s[mrow * LP + k + 1];
        }
        #pragma unroll
        for (int nt = 0; nt < 4; ++nt) {
            v16h bfrag;
            int nrow = nt * 16 + (lane & 15);
            #pragma unroll
            for (int e = 0; e < 16; ++e) {
                bfrag[e] = wf16_s[nrow * WP + ks * 32 + koffB + (e & 7) + ((e >> 3) << 4)];
            }
            acc[nt] = __builtin_amdgcn_wmma_f32_16x16x32_f16(
                false, afrag, false, bfrag, (short)0, acc[nt], false, false);
        }
    }

    // -------- epilogue: gated = item * sigmoid(S + gadd) --------
    #pragma unroll
    for (int nt = 0; nt < 4; ++nt) {
        #pragma unroll
        for (int r = 0; r < 8; ++r) {
            int l = wave * 16 + r + ((lane >> 4) << 3);
            int d = nt * 16 + (lane & 15);
            float x = acc[nt][r] + gadd_s[d];
            gated_s[l * LP + d] = items_s[l * LP + d] * sigmoidf(x);
        }
    }
    __syncthreads();

    // -------- instance gate: inst[l] = sigmoid(gated.ig_item + user.ig_user[:,l]) --------
    if (tid < L) {
        float t1 = 0.f, t2 = 0.f;
        #pragma unroll 8
        for (int k = 0; k < 64; ++k) {
            t1 += gated_s[tid * LP + k] * inst_gate_item[k];
            t2 += user_s[k] * inst_gate_user[k * L + tid];
        }
        inst_s[tid] = sigmoidf(t1 + t2);
    }
    __syncthreads();

    // -------- v[b] = user + union + sum_l items --------
    if (tid < 64) {
        float un = 0.f, isum = 0.f, itsum = 0.f;
        for (int l = 0; l < L; ++l) {
            float w = inst_s[l];
            un    += gated_s[l * LP + tid] * w;
            isum  += w;
            itsum += items_s[l * LP + tid];
        }
        v_out[(size_t)b * DD + tid] = user_s[tid] + un / isum + itsum;
    }
}

// One block per batch row, 16 lanes per target: res[b,t] = b2[id] + dot(w2[id], v[b])
__global__ __launch_bounds__(256) void hgn_score_kernel(
    const int*   __restrict__ target_ids,  // [B,T]
    const float* __restrict__ W2_table,    // [I,64]
    const float* __restrict__ b2_table,    // [I,1]
    const float* __restrict__ v_in,        // [B,64]
    float*       __restrict__ out,         // [B,T]
    int T)
{
    __shared__ __align__(16) float4 vv[16];
    const int b   = blockIdx.x;
    const int tid = threadIdx.x;
    if (tid < 16) vv[tid] = ((const float4*)(v_in + (size_t)b * DD))[tid];
    __syncthreads();

    const int q  = tid & 15;   // float4 index within the 64-float row (0..15)
    const int tg = tid >> 4;   // target slot (0..15)
    const float4 x = vv[q];

    for (int t = tg; t < T; t += 16) {
        int id = target_ids[(size_t)b * T + t];
        int tn = t + 16;
        if (tn < T) {
            int idn = target_ids[(size_t)b * T + tn];
            __builtin_prefetch(W2_table + (size_t)idn * DD + q * 4, 0, 0);
        }
        float4 w = ((const float4*)(W2_table + (size_t)id * DD))[q];
        float s = w.x * x.x + w.y * x.y + w.z * x.z + w.w * x.w;
        s += __shfl_xor(s, 1);
        s += __shfl_xor(s, 2);
        s += __shfl_xor(s, 4);
        s += __shfl_xor(s, 8);
        if (q == 0) out[(size_t)b * T + t] = s + b2_table[id];
    }
}

extern "C" void kernel_launch(void* const* d_in, const int* in_sizes, int n_in,
                              void* d_out, int out_size, void* d_ws, size_t ws_size,
                              hipStream_t stream) {
    const int*   user_ids     = (const int*)  d_in[0];
    const int*   item_seq_ids = (const int*)  d_in[1];
    const int*   target_ids   = (const int*)  d_in[2];
    const float* user_table   = (const float*)d_in[3];
    const float* item_table   = (const float*)d_in[4];
    const float* W2_table     = (const float*)d_in[5];
    const float* b2_table     = (const float*)d_in[6];
    const float* fg_item_W    = (const float*)d_in[7];
    const float* fg_item_b    = (const float*)d_in[8];
    const float* fg_user_W    = (const float*)d_in[9];
    const float* fg_user_b    = (const float*)d_in[10];
    const float* inst_g_item  = (const float*)d_in[11];
    const float* inst_g_user  = (const float*)d_in[12];

    const int B = in_sizes[0];
    const int L = in_sizes[1] / B;
    const int T = in_sizes[2] / B;
    const int num_items = in_sizes[4] / DD;

    float* v_ws = (float*)d_ws;  // [B,64] fp32 scratch

    hgn_user_kernel<<<B, 128, 0, stream>>>(
        user_ids, item_seq_ids, user_table, item_table,
        fg_item_W, fg_item_b, fg_user_W, fg_user_b,
        inst_g_item, inst_g_user, v_ws, L, num_items);

    hgn_score_kernel<<<B, 256, 0, stream>>>(
        target_ids, W2_table, b2_table, v_ws, (float*)d_out, T);
}
